// GAT_56556129353738
// MI455X (gfx1250) — matrix operand order repaired and checked by reading, not compiled, not run on hardware
//
#include <hip/hip_runtime.h>
#include <hip/hip_bf16.h>
#include <math.h>

// ---------------------------------------------------------------------------
// GAT layer for MI455X (gfx1250, wave32).
//   k1: W fp32 -> bf16 (once per call, 16K elems)
//   k2: edge-parallel scores: one wave32 per edge, float4/lane, shfl_xor reduce
//   k3: CSR row offsets via binary search on sorted dst_idx (no atomics)
//   k4: per-dst-node softmax + weighted gather (128 thr = 1 feature/thread),
//       writes rst directly as bf16
//   k5: out = relu(rst @ W^T + b) via v_wmma_f32_16x16x32_bf16, W staged in LDS
// ---------------------------------------------------------------------------

typedef __attribute__((ext_vector_type(8)))  float  v8f;
typedef __attribute__((ext_vector_type(8)))  __bf16 v8bf;
typedef __attribute__((ext_vector_type(16))) __bf16 v16bf;

#define D_FEAT 128

// ---------------- k1: convert W to bf16 ----------------
__global__ void k_convert_w(const float* __restrict__ W, __bf16* __restrict__ Wb,
                            int total) {
  int i = blockIdx.x * blockDim.x + threadIdx.x;
  if (i < total) Wb[i] = (__bf16)W[i];
}

// ---------------- k2: per-edge dot-product scores ----------------
// One wave32 per edge; each lane handles 4 of the 128 features (float4).
__global__ void __launch_bounds__(256) k_score(
    const float* __restrict__ fsrc, const float* __restrict__ fdst,
    const int* __restrict__ si, const int* __restrict__ di,
    float* __restrict__ score, int E) {
  int lane = threadIdx.x & 31;
  int eid  = blockIdx.x * (blockDim.x >> 5) + (threadIdx.x >> 5);
  if (eid >= E) return;
  int s = si[eid], d = di[eid];
  float4 a = ((const float4*)(fsrc + (size_t)s * D_FEAT))[lane];
  float4 b = ((const float4*)(fdst + (size_t)d * D_FEAT))[lane];
  float dot = a.x * b.x + a.y * b.y + a.z * b.z + a.w * b.w;
#pragma unroll
  for (int off = 16; off > 0; off >>= 1) dot += __shfl_xor(dot, off, 32);
  if (lane == 0) score[eid] = dot;
}

// ---------------- k3: CSR row offsets (dst_idx is sorted) ----------------
__global__ void k_rowptr(const int* __restrict__ di, int* __restrict__ rp,
                         int N, int E) {
  int n = blockIdx.x * blockDim.x + threadIdx.x;
  if (n > N) return;
  int lo = 0, hi = E;            // lower_bound(di, n); rp[N] == E
  while (lo < hi) {
    int mid = (lo + hi) >> 1;
    if (di[mid] < n) lo = mid + 1; else hi = mid;
  }
  rp[n] = lo;
}

// ---------------- k4: per-node edge softmax + weighted gather ----------------
// One 128-thread block per dst node; thread t owns feature dim t, so the
// feat_src gather is a fully coalesced 512B read per edge. No atomics.
__global__ void __launch_bounds__(128) k_aggregate(
    const float* __restrict__ fsrc, const int* __restrict__ si,
    const float* __restrict__ score, const int* __restrict__ rp,
    __bf16* __restrict__ rstb, int N) {
  int n = blockIdx.x;
  int t = threadIdx.x;
  int beg = rp[n], end = rp[n + 1];
  float acc = 0.0f;
  if (beg < end) {
    float smax = -INFINITY;
    for (int e = beg; e < end; ++e) smax = fmaxf(smax, score[e]);
    float denom = 0.0f;
    for (int e = beg; e < end; ++e) denom += __expf(score[e] - smax);
    float inv = 1.0f / denom;
    for (int e = beg; e < end; ++e) {
      float w = __expf(score[e] - smax) * inv;  // score[] broadcast-hits L1/L2
      int s = si[e];
      acc += w * fsrc[(size_t)s * D_FEAT + t];
    }
  }
  rstb[(size_t)n * D_FEAT + t] = (__bf16)acc;   // rst stored bf16: halves re-read
}

// ---------------- k5: out = relu(rst @ W^T + b), WMMA bf16 ----------------
// 256 threads = 8 waves per block; each wave computes a 16x128 output strip.
// W (bf16, 32KB) is staged once per block in LDS and shared by all 8 waves.
__global__ void __launch_bounds__(256) k_gemm(
    const __bf16* __restrict__ rstb, const __bf16* __restrict__ Wb,
    const float* __restrict__ bias, float* __restrict__ out, int N) {
  __shared__ __bf16 Wl[D_FEAT * D_FEAT];        // 32 KB of 320 KB/WGP
  {
    const uint4* g = (const uint4*)Wb;
    uint4*       s = (uint4*)Wl;
    for (int i = threadIdx.x; i < (D_FEAT * D_FEAT * 2) / 16; i += 256)
      s[i] = g[i];
  }
  __syncthreads();

  int lane = threadIdx.x & 31;
  int wave = threadIdx.x >> 5;
  long Mbase = ((long)blockIdx.x * 8 + wave) * 16;
  if (Mbase >= N) return;                       // N is a multiple of 16

  int mrow  = lane & 15;
  int half  = lane >> 4;
  int koffA = half * 16;                        // A: lane<16 -> K 0..15 (contig)
  int koffB = half * 16;                        // B: lane<16 -> K 0..15 (contig)

  const __bf16* arow = rstb + (Mbase + mrow) * D_FEAT;

  v8f acc[8] = {};                              // 8 N-tiles of 16 cols each
#pragma unroll
  for (int k0 = 0; k0 < D_FEAT; k0 += 32) {
    // A fragment: 16x32 bf16, two 16B chunks per lane (contiguous K order, same as B)
    v8bf alo = *(const v8bf*)(arow + k0 + koffA);
    v8bf ahi = *(const v8bf*)(arow + k0 + koffA + 8);
    v16bf a = __builtin_shufflevector(alo, ahi, 0, 1, 2, 3, 4, 5, 6, 7,
                                      8, 9, 10, 11, 12, 13, 14, 15);
#pragma unroll
    for (int nt = 0; nt < 8; ++nt) {
      // B column (nt*16 + mrow) == W row (out col), contiguous along K in LDS
      const __bf16* bcol = Wl + (nt * 16 + mrow) * D_FEAT;
      v16bf b = *(const v16bf*)(bcol + k0 + koffB);   // 32B ds_load
      acc[nt] = __builtin_amdgcn_wmma_f32_16x16x32_bf16(
          false, a, false, b, (short)0, acc[nt], false, false);
    }
  }

  // D layout: lane l -> col N = l&15; VGPR r -> row M = r + 8*(l>>4)
  int ncol  = lane & 15;
  int rbase = half * 8;
#pragma unroll
  for (int nt = 0; nt < 8; ++nt) {
    int o = nt * 16 + ncol;
    float bv = bias[o];
#pragma unroll
    for (int r = 0; r < 8; ++r) {
      float v = acc[nt][r] + bv;
      out[(Mbase + rbase + r) * D_FEAT + o] = v > 0.0f ? v : 0.0f;
    }
  }
}

// ---------------------------------------------------------------------------
extern "C" void kernel_launch(void* const* d_in, const int* in_sizes, int n_in,
                              void* d_out, int out_size, void* d_ws,
                              size_t ws_size, hipStream_t stream) {
  const float* fsrc = (const float*)d_in[0];
  const float* fdst = (const float*)d_in[1];
  const float* W    = (const float*)d_in[2];
  const float* b    = (const float*)d_in[3];
  const int*   si   = (const int*)d_in[4];
  const int*   di   = (const int*)d_in[5];
  float*       out  = (float*)d_out;

  const int N = in_sizes[0] / D_FEAT;   // 100000
  const int E = in_sizes[4];            // 1600000

  // Workspace layout (all 16B-aligned): rst_bf16 | W_bf16 | score | row_start
  char* ws = (char*)d_ws;
  __bf16* rstb  = (__bf16*)ws;                    size_t off = (size_t)N * D_FEAT * 2;
  __bf16* Wb    = (__bf16*)(ws + off);            off += (size_t)D_FEAT * D_FEAT * 2;
  float*  score = (float*)(ws + off);             off += (size_t)E * 4;
  int*    rp    = (int*)(ws + off);

  // k1: W -> bf16
  k_convert_w<<<(D_FEAT * D_FEAT + 255) / 256, 256, 0, stream>>>(
      W, Wb, D_FEAT * D_FEAT);

  // k2: edge scores (8 edges / 256-thread block)
  k_score<<<(E + 7) / 8, 256, 0, stream>>>(fsrc, fdst, si, di, score, E);

  // k3: row offsets
  k_rowptr<<<(N + 1 + 255) / 256, 256, 0, stream>>>(di, rp, N, E);

  // k4: softmax + weighted gather (one block per dst node)
  k_aggregate<<<N, 128, 0, stream>>>(fsrc, si, score, rp, rstb, N);

  // k5: final linear + ReLU via WMMA (128 rows per 256-thread block)
  k_gemm<<<(N + 127) / 128, 256, 0, stream>>>(rstb, Wb, b, out, N);
}
